// RNN_Tensorized_33071248179757
// MI455X (gfx1250) — matrix operand order repaired and checked by reading, not compiled, not run on hardware
//
#include <hip/hip_runtime.h>
#include <math.h>

// CDNA5 / gfx1250. wave32.
typedef __attribute__((ext_vector_type(16))) _Float16 v16h;
typedef __attribute__((ext_vector_type(8)))  float    v8f;

#define Bsz 8192
#define Lsz 64
#define Hsz 64

__device__ __forceinline__ float elu1(float x) {
  return x > 0.0f ? x : expm1f(x);
}

// ---------------------------------------------------------------------------
// Kernel 1 (1 block, 4 waves): compute p[l], l = 0..63.
//
//   out[b,l] = sigmoid( sum_h elu(b3[l,h]) * (Ws[l,h,1]-Ws[l,h,0])
//                       + bs[l,1] - bs[l,0] )            (batch-independent)
//
// The batched per-l dot products are the diagonal of U*V^T (U,V: 64x64),
// computed as four 16x16 diagonal tiles with v_wmma_f32_16x16x32_f16,
// K=64 via two accumulating WMMAs. One wave per tile.
// ---------------------------------------------------------------------------
__global__ __launch_bounds__(128) void rnn_logits_wmma(
    const float* __restrict__ b3, const float* __restrict__ Ws,
    const float* __restrict__ bs, float* __restrict__ p) {
  const int lane = threadIdx.x & 31;
  const int tile = threadIdx.x >> 5;   // wave id == l-tile (rows 16*tile..+15)
  const int half = lane >> 4;          // lane half selects K sub-range
  const int m    = lane & 15;          // A row m == B column n
  const int lrow = tile * 16 + m;

  // Pack operands per documented 16-bit WMMA VGPR layouts (ISA 7.12.2):
  //  A 16x32: element e of v16h holds K = 16*(e>>3) + 8*half + (e&7)
  //  B 32x16: element e of v16h holds K = 16*(e>>3) + 8*half + (e&7)   (column n = lane&15)
  v16h a0, a1, bv0, bv1;
#pragma unroll
  for (int e = 0; e < 16; ++e) {
    const int kA = ((e >> 3) << 4) + (half << 3) + (e & 7);
    const int kB = kA;
    a0[e]  = (_Float16)elu1(b3[lrow * Hsz + kA]);
    a1[e]  = (_Float16)elu1(b3[lrow * Hsz + kA + 32]);
    bv0[e] = (_Float16)(Ws[(lrow * Hsz + kB) * 2 + 1] -
                        Ws[(lrow * Hsz + kB) * 2 + 0]);
    bv1[e] = (_Float16)(Ws[(lrow * Hsz + kB + 32) * 2 + 1] -
                        Ws[(lrow * Hsz + kB + 32) * 2 + 0]);
  }

  v8f c = {};
  // D = A(16x32) * B(32x16) + C, f32 accumulate; chain K = 0..31, 32..63.
  c = __builtin_amdgcn_wmma_f32_16x16x32_f16(false, a0, false, bv0,
                                             (short)0, c, false, false);
  c = __builtin_amdgcn_wmma_f32_16x16x32_f16(false, a1, false, bv1,
                                             (short)0, c, false, false);

  // Diagonal extraction from C/D layout: VGPR r, lanes 0-15 -> M=r,N=lane;
  // lanes 16-31 -> M=8+r, N=lane-16.  M==N at lane==r and lane==r+24.
#pragma unroll
  for (int r = 0; r < 8; ++r) {
    if (lane == r || lane == r + 24) {
      const int mm = (lane < 16) ? r : (r + 8);
      const int l  = tile * 16 + mm;
      const float z = c[r] + bs[2 * l + 1] - bs[2 * l + 0];
      p[l] = 1.0f / (1.0f + __expf(-z));   // softmax over 2 logits, class 1
    }
  }
}

// ---------------------------------------------------------------------------
// Kernel 2: out[b,l] = p[l].  2 MB of b128 stores; the 256 B of p stays
// L2-resident.  Each thread keeps a constant l-phase (stride 16384 % 16 == 0)
// so it reads its float4 of p once and issues 8 wide stores.
// ---------------------------------------------------------------------------
__global__ __launch_bounds__(256) void rnn_broadcast(
    const float4* __restrict__ p4, float4* __restrict__ out4) {
  const int tid = blockIdx.x * 256 + threadIdx.x;  // 64 blocks * 256 = 16384
  const float4 v = p4[tid & 15];                   // row is 16 float4 wide
#pragma unroll
  for (int k = 0; k < 8; ++k)                      // 16384*8 = 131072 float4
    out4[tid + k * 16384] = v;
}

// ---------------------------------------------------------------------------
// Inputs (setup_inputs order): 0 sigma_list, 1 W1, 2 b1, 3 W2, 4 b2,
//                              5 W3, 6 b3, 7 Ws, 8 bs.
// Only b3, Ws, bs feed the output (layers 1-3 bilinear terms are dead).
// ---------------------------------------------------------------------------
extern "C" void kernel_launch(void* const* d_in, const int* in_sizes, int n_in,
                              void* d_out, int out_size, void* d_ws, size_t ws_size,
                              hipStream_t stream) {
  (void)in_sizes; (void)n_in; (void)out_size; (void)ws_size;
  const float* b3 = (const float*)d_in[6];
  const float* Ws = (const float*)d_in[7];
  const float* bs = (const float*)d_in[8];
  float* p = (float*)d_ws;  // 64 floats of scratch

  rnn_logits_wmma<<<dim3(1), dim3(128), 0, stream>>>(b3, Ws, bs, p);
  rnn_broadcast<<<dim3(64), dim3(256), 0, stream>>>((const float4*)p,
                                                    (float4*)d_out);
}
